// MultiHeadAttention_1065151889488
// MI455X (gfx1250) — matrix operand order repaired and checked by reading, not compiled, not run on hardware
//
#include <hip/hip_runtime.h>

// ---------------------------------------------------------------------------
// MI455X (gfx1250) multi-head "batch-axis" attention, bf16 WMMA pipeline.
//   1) convert W{q,k,v,o} fp32 -> bf16 (once; reused by every GEMM wave)
//   2) q/k/v = X @ W^T + b  (WMMA bf16, f32 accumulate, bf16 out)
//   3) flash attention per (t,h) over the batch axis (online softmax,
//      K tile staged with gfx1250 async global->LDS DMA)
//   4) out = O @ Wo^T + bo  (fp32 out)
// ---------------------------------------------------------------------------

typedef __bf16 bf16;
typedef __attribute__((ext_vector_type(16))) __bf16 bf16x16;
typedef __attribute__((ext_vector_type(8)))  __bf16 bf16x8;
typedef __attribute__((ext_vector_type(4)))  __bf16 bf16x4;
typedef __attribute__((ext_vector_type(8)))  float  f32x8;
typedef __attribute__((ext_vector_type(4)))  float  f32x4;

#define EMBED 512
#define HEADS 8
#define HDIM  64
#define NBAT  512
#define TSEQ  64
#define ROWS  (NBAT * TSEQ)   // 32768

// native conversion: backend selects v_cvt_pk_bf16_f32 on gfx1250
__device__ __forceinline__ bf16 f2bf(float f) { return (bf16)f; }

__device__ __forceinline__ f32x8 wmma_bf16(bf16x16 a, bf16x16 b, f32x8 c) {
  // D = A(16x32 bf16) * B(32x16 bf16) + C(16x16 f32)
  return __builtin_amdgcn_wmma_f32_16x16x32_bf16(false, a, false, b, (short)0, c,
                                                 false, false);
}

// gfx1250 async DMA: 16 bytes global -> LDS per lane, tracked by ASYNCcnt.
// Generic LDS pointers carry the LDS byte offset in their low 32 bits.
__device__ __forceinline__ void async_g2l_b128(void* lds, const void* gptr) {
  unsigned lds_off = (unsigned)(size_t)lds;
  asm volatile("global_load_async_to_lds_b128 %0, %1, off"
               :: "v"(lds_off), "v"(gptr)
               : "memory");
}
__device__ __forceinline__ void wait_asynccnt0() {
  asm volatile("s_wait_asynccnt 0x0" ::: "memory");
}

// ---------------------------------------------------------------------------
// Kernel 0: bulk fp32 -> bf16 conversion (vectorized, grid-stride)
// ---------------------------------------------------------------------------
__global__ __launch_bounds__(256) void cvt_bf16(const float* __restrict__ src,
                                                bf16* __restrict__ dst, int n4) {
  int i = blockIdx.x * blockDim.x + threadIdx.x;
  int stride = gridDim.x * blockDim.x;
  for (; i < n4; i += stride) {
    f32x4 v = ((const f32x4*)src)[i];
    bf16x4 r;
    r[0] = f2bf(v[0]); r[1] = f2bf(v[1]); r[2] = f2bf(v[2]); r[3] = f2bf(v[3]);
    ((bf16x4*)dst)[i] = r;
  }
}

// ---------------------------------------------------------------------------
// Kernel 1: Y(bf16, MxE) = X(f32, MxE) @ Wb(bf16, ExE)^T + bias
// One wave computes a 16x64 output tile; block = 8 waves = 128 rows.
// grid.x = column block (fast-varying -> L2 reuse of the X slab), grid.y = row.
// ---------------------------------------------------------------------------
__global__ __launch_bounds__(256) void proj_gemm(const float* __restrict__ X,
                                                 const bf16* __restrict__ Wb,
                                                 const float* __restrict__ bias,
                                                 bf16* __restrict__ Y) {
  const int lane = threadIdx.x & 31;
  const int wv   = threadIdx.x >> 5;
  const int half = lane >> 4;
  const int l16  = lane & 15;
  const int n0 = blockIdx.x * 64;
  const int m0 = blockIdx.y * 128 + wv * 16;

  f32x8 acc[4] = {};
  const float* xrow = X + (size_t)(m0 + l16) * EMBED;

  for (int k = 0; k < EMBED; k += 32) {
    // A fragment: lane holds row m0+l16; elems 0-7 -> K=k+16*half+i,
    // elems 8-15 -> K=k+16*half+8+i  (contiguous, matches B)
    bf16x16 a;
    {
      const f32x4* p0 = (const f32x4*)(xrow + k + 16 * half);
      const f32x4* p1 = (const f32x4*)(xrow + k + 16 * half + 8);
      f32x4 lo0 = p0[0], lo1 = p0[1];
      f32x4 hi0 = p1[0], hi1 = p1[1];
#pragma unroll
      for (int i = 0; i < 4; ++i) {
        a[i]      = f2bf(lo0[i]);
        a[4 + i]  = f2bf(lo1[i]);
        a[8 + i]  = f2bf(hi0[i]);
        a[12 + i] = f2bf(hi1[i]);
      }
    }
#pragma unroll
    for (int j = 0; j < 4; ++j) {
      // B fragment: B[ki][n] = W[n0+16j+n][k+ki]; lane elem jj -> ki=16*half+jj
      bf16x16 b = *(const bf16x16*)(Wb + (size_t)(n0 + 16 * j + l16) * EMBED +
                                    k + 16 * half);
      acc[j] = wmma_bf16(a, b, acc[j]);
    }
  }

  // C layout: VGPR r, lanes 0-15 -> row r, lanes 16-31 -> row r+8; col = l16
#pragma unroll
  for (int j = 0; j < 4; ++j) {
    const int col = n0 + 16 * j + l16;
    const float bcol = bias[col];
#pragma unroll
    for (int r = 0; r < 8; ++r) {
      const int row = m0 + r + 8 * half;
      Y[(size_t)row * EMBED + col] = f2bf(acc[j][r] + bcol);
    }
  }
}

// ---------------------------------------------------------------------------
// Kernel 2: flash attention over the batch axis, per (t, h).
// Block: 128 query rows x one (t,h); 8 waves; streams K/V in 32-row tiles.
// K tile: async global->LDS DMA (ASYNCcnt).  V tile: manual transpose.
// ---------------------------------------------------------------------------
__global__ __launch_bounds__(256) void attn_kernel(const bf16* __restrict__ Q,
                                                   const bf16* __restrict__ K,
                                                   const bf16* __restrict__ V,
                                                   bf16* __restrict__ O) {
  __shared__ __align__(32) bf16 Ks[32][64];       // K tile, row-major
  __shared__ __align__(32) bf16 Vt[64][32];       // V tile, transposed [d][n]
  __shared__ __align__(16) bf16 Ps[8][16][32];    // per-wave P relayout scratch

  const int lane = threadIdx.x & 31;
  const int wv   = threadIdx.x >> 5;
  const int half = lane >> 4;
  const int l16  = lane & 15;
  const int t = blockIdx.y, h = blockIdx.z;
  const int q0 = blockIdx.x * 128 + wv * 16;
  const size_t headoff = (size_t)h * HDIM;

  // per-thread cooperative-copy coordinates (fixed across iterations)
  const int tid = threadIdx.x;
  const int kr = tid >> 3, kc = (tid & 7) * 8;     // K tile: 32 rows x 8 chunks
  const int vr = tid & 31, vc = (tid >> 5) * 8;    // V tile: 32 rows x 8 chunks
  const bf16* kbase = K + ((size_t)kr * TSEQ + t) * EMBED + headoff + kc;
  const bf16* vbase = V + ((size_t)vr * TSEQ + t) * EMBED + headoff + vc;
  const size_t tilestep = (size_t)32 * TSEQ * EMBED;  // 32 batch rows

  // Q fragments: 16 rows x 64 head-dims = two K=32 A-fragments
  bf16x16 qf[2];
  {
    const size_t qrow = ((size_t)(q0 + l16) * TSEQ + t) * EMBED + headoff;
#pragma unroll
    for (int c = 0; c < 2; ++c) {
      bf16x8 lo = *(const bf16x8*)(Q + qrow + 32 * c + 16 * half);
      bf16x8 hi = *(const bf16x8*)(Q + qrow + 32 * c + 16 * half + 8);
#pragma unroll
      for (int i = 0; i < 8; ++i) { qf[c][i] = lo[i]; qf[c][8 + i] = hi[i]; }
    }
  }

  f32x8 o[4] = {};
  float mrow[8], lrow[8];
#pragma unroll
  for (int r = 0; r < 8; ++r) { mrow[r] = -3.0e38f; lrow[r] = 0.0f; }

  const float scale = 0.044194173824159216f;  // 1/sqrt(EMBED)

  for (int it = 0; it < NBAT / 32; ++it) {
    // ---- stage tiles: K via async DMA to LDS, V via transposing copy ----
    async_g2l_b128(&Ks[kr][kc], kbase + (size_t)it * tilestep);
    {
      bf16x8 vvec = *(const bf16x8*)(vbase + (size_t)it * tilestep);
#pragma unroll
      for (int i = 0; i < 8; ++i) Vt[vc + i][vr] = vvec[i];
    }
    // prefetch next iteration's K/V lines while this tile is computed on
    if (it + 1 < NBAT / 32) {
      __builtin_prefetch(kbase + (size_t)(it + 1) * tilestep, 0, 1);
      __builtin_prefetch(vbase + (size_t)(it + 1) * tilestep, 0, 1);
    }
    wait_asynccnt0();   // our async K chunk has landed in LDS
    __syncthreads();    // everyone's chunks visible

    // ---- S(16x32) = Q(16x64) @ Kb^T : two 16x16 column tiles ----
    f32x8 s[2] = {};
#pragma unroll
    for (int j = 0; j < 2; ++j) {
#pragma unroll
      for (int c = 0; c < 2; ++c) {
        bf16x16 b = *(const bf16x16*)&Ks[16 * j + l16][32 * c + 16 * half];
        s[j] = wmma_bf16(qf[c], b, s[j]);
      }
    }

    // ---- online softmax over the 32 new columns ----
    float alpha[8];
#pragma unroll
    for (int r = 0; r < 8; ++r) {
      float s0 = s[0][r] * scale;
      float s1 = s[1][r] * scale;
      float rmax = fmaxf(s0, s1);
#pragma unroll
      for (int off = 1; off < 16; off <<= 1)
        rmax = fmaxf(rmax, __shfl_xor(rmax, off, 32));  // within 16-lane half
      float mnew = fmaxf(mrow[r], rmax);
      float p0 = __expf(s0 - mnew);
      float p1 = __expf(s1 - mnew);
      float rs = p0 + p1;
#pragma unroll
      for (int off = 1; off < 16; off <<= 1) rs += __shfl_xor(rs, off, 32);
      alpha[r] = __expf(mrow[r] - mnew);
      lrow[r] = lrow[r] * alpha[r] + rs;
      mrow[r] = mnew;
      Ps[wv][r + 8 * half][l16]      = f2bf(p0);
      Ps[wv][r + 8 * half][16 + l16] = f2bf(p1);
    }
#pragma unroll
    for (int j = 0; j < 4; ++j)
#pragma unroll
      for (int r = 0; r < 8; ++r) o[j][r] *= alpha[r];

    __syncthreads();  // Ps C-layout writes -> A-layout reads

    // ---- P re-layout to A fragment ----
    bf16x16 pf;
    {
      bf16x8 lo = *(const bf16x8*)&Ps[wv][l16][16 * half];
      bf16x8 hi = *(const bf16x8*)&Ps[wv][l16][16 * half + 8];
#pragma unroll
      for (int i = 0; i < 8; ++i) { pf[i] = lo[i]; pf[8 + i] = hi[i]; }
    }

    // ---- O(16x64) += P(16x32) @ Vb(32x64) ----
#pragma unroll
    for (int j = 0; j < 4; ++j) {
      bf16x16 b = *(const bf16x16*)&Vt[16 * j + l16][16 * half];
      o[j] = wmma_bf16(pf, b, o[j]);
    }
    __syncthreads();  // protect Ks/Vt/Ps before next tile
  }

  // ---- epilogue: normalize rows, store bf16 at [n][t][h*64+d] ----
#pragma unroll
  for (int r = 0; r < 8; ++r) {
    const float inv = 1.0f / lrow[r];
    const size_t orow = ((size_t)(q0 + r + 8 * half) * TSEQ + t) * EMBED + headoff;
#pragma unroll
    for (int j = 0; j < 4; ++j) O[orow + 16 * j + l16] = f2bf(o[j][r] * inv);
  }
}

// ---------------------------------------------------------------------------
// Kernel 3: out(f32) = A(bf16, MxE) @ Wb(bf16, ExE)^T + bo
// ---------------------------------------------------------------------------
__global__ __launch_bounds__(256) void out_gemm(const bf16* __restrict__ A,
                                                const bf16* __restrict__ Wb,
                                                const float* __restrict__ bias,
                                                float* __restrict__ Y) {
  const int lane = threadIdx.x & 31;
  const int wv   = threadIdx.x >> 5;
  const int half = lane >> 4;
  const int l16  = lane & 15;
  const int n0 = blockIdx.x * 64;
  const int m0 = blockIdx.y * 128 + wv * 16;

  f32x8 acc[4] = {};
  const bf16* arow = A + (size_t)(m0 + l16) * EMBED;

  for (int k = 0; k < EMBED; k += 32) {
    bf16x16 a;
    {
      bf16x8 lo = *(const bf16x8*)(arow + k + 16 * half);
      bf16x8 hi = *(const bf16x8*)(arow + k + 16 * half + 8);
#pragma unroll
      for (int i = 0; i < 8; ++i) { a[i] = lo[i]; a[8 + i] = hi[i]; }
    }
#pragma unroll
    for (int j = 0; j < 4; ++j) {
      bf16x16 b = *(const bf16x16*)(Wb + (size_t)(n0 + 16 * j + l16) * EMBED +
                                    k + 16 * half);
      acc[j] = wmma_bf16(a, b, acc[j]);
    }
  }

#pragma unroll
  for (int j = 0; j < 4; ++j) {
    const int col = n0 + 16 * j + l16;
    const float bcol = bias[col];
#pragma unroll
    for (int r = 0; r < 8; ++r) {
      const int row = m0 + r + 8 * half;
      Y[(size_t)row * EMBED + col] = acc[j][r] + bcol;
    }
  }
}

// ---------------------------------------------------------------------------
extern "C" void kernel_launch(void* const* d_in, const int* in_sizes, int n_in,
                              void* d_out, int out_size, void* d_ws, size_t ws_size,
                              hipStream_t stream) {
  const float* values = (const float*)d_in[0];
  const float* keys   = (const float*)d_in[1];
  const float* query  = (const float*)d_in[2];
  const float* Wv = (const float*)d_in[3];
  const float* bv = (const float*)d_in[4];
  const float* Wk = (const float*)d_in[5];
  const float* bk = (const float*)d_in[6];
  const float* Wq = (const float*)d_in[7];
  const float* bq = (const float*)d_in[8];
  const float* Wo = (const float*)d_in[9];
  const float* bo = (const float*)d_in[10];

  // workspace: q/k/v projections + attention output (bf16 [ROWS][EMBED]),
  // then 4 bf16 weight matrices.
  bf16* qp  = (bf16*)d_ws;
  bf16* kp  = qp + (size_t)ROWS * EMBED;
  bf16* vp  = kp + (size_t)ROWS * EMBED;
  bf16* op  = vp + (size_t)ROWS * EMBED;
  bf16* Wqb = op + (size_t)ROWS * EMBED;
  bf16* Wkb = Wqb + (size_t)EMBED * EMBED;
  bf16* Wvb = Wkb + (size_t)EMBED * EMBED;
  bf16* Wob = Wvb + (size_t)EMBED * EMBED;

  dim3 blk(256);

  // weight conversion: 512*512/4 = 65536 vec4 per matrix
  const int wn4 = EMBED * EMBED / 4;
  cvt_bf16<<<dim3(64), blk, 0, stream>>>(Wq, Wqb, wn4);
  cvt_bf16<<<dim3(64), blk, 0, stream>>>(Wk, Wkb, wn4);
  cvt_bf16<<<dim3(64), blk, 0, stream>>>(Wv, Wvb, wn4);
  cvt_bf16<<<dim3(64), blk, 0, stream>>>(Wo, Wob, wn4);

  dim3 gemm_grid(EMBED / 64, ROWS / 128);  // col-block fastest for L2 reuse
  proj_gemm<<<gemm_grid, blk, 0, stream>>>(query,  Wqb, bq, qp);
  proj_gemm<<<gemm_grid, blk, 0, stream>>>(keys,   Wkb, bk, kp);
  proj_gemm<<<gemm_grid, blk, 0, stream>>>(values, Wvb, bv, vp);

  dim3 attn_grid(NBAT / 128, TSEQ, HEADS);
  attn_kernel<<<attn_grid, blk, 0, stream>>>(qp, kp, vp, op);

  out_gemm<<<gemm_grid, blk, 0, stream>>>(op, Wob, bo, (float*)d_out);
}
